// MixtureOfExpertsLayer_53395033424483
// MI455X (gfx1250) — matrix operand order repaired and checked by reading, not compiled, not run on hardware
//
#include <hip/hip_runtime.h>

#define N_TOK 8192
#define DIM   1024
#define UDIM  512
#define NEXP  8
#define NBASE 8
#define MT    32                         // tokens per tile
#define MAXT  (N_TOK / MT + NEXP)        // 264 tiles max (per-expert padding)
#define KB1   (DIM / 32)                 // 32 K-blocks, GEMM1
#define KB2   (UDIM / 32)                // 16 K-blocks, GEMM2
#define NCT   (UDIM / 16)                // 32 column tiles
#define KSTR  (NCT * 32 * 16)            // ushorts per K-block of swizzled weights

typedef __bf16 v16bf __attribute__((ext_vector_type(16)));
typedef float  v8f   __attribute__((ext_vector_type(8)));

static __device__ __forceinline__ unsigned short f32_to_bf16(float f) {
  unsigned int u = __float_as_uint(f);
  u += 0x7FFFu + ((u >> 16) & 1u);       // round-to-nearest-even
  return (unsigned short)(u >> 16);
}

// ------------------------------------------------------------ init routing state
__global__ void init_kernel(int* counts, int* cursors, int* perm) {
  int i = blockIdx.x * blockDim.x + threadIdx.x;
  if (i < NEXP) { counts[i] = 0; cursors[i] = 0; }
  if (i < MAXT * MT) perm[i] = -1;
}

// ------------------------------------------------------------ one-time weight convert+swizzle
// Destination layout = WMMA-B fragment tiles:
//   frag(e,kb,ct): 32 lanes x 16 bf16 contiguous; lane j holds col ct*16+(j&15),
//   K = kb*32 + 8*((j>>4)&1) + (s&7) + 16*(s>>3)  (s = 0..15)
__global__ __launch_bounds__(256) void wconv_kernel(
    const float* __restrict__ W1, const float* __restrict__ proj,
    unsigned short* __restrict__ W1b, unsigned short* __restrict__ projb) {
  int gid = blockIdx.x * 256 + threadIdx.x;          // one float4 each
  const int n1 = NEXP * DIM * UDIM / 4;              // 1M float4
  const int n2 = NEXP * UDIM * UDIM / 4;             // 512K float4
  if (gid < n1) {
    int idx = gid * 4;
    int u = idx % UDIM; int rest = idx / UDIM;
    int k = rest % DIM; int e = rest / DIM;
    float4 v = *(const float4*)(W1 + (size_t)idx);
    int kb = k >> 5, kr = k & 31;
    int s = (kr & 7) + ((kr >> 4) << 3), lhalf = ((kr >> 3) & 1) << 4;
    float vv[4] = {v.x, v.y, v.z, v.w};
#pragma unroll
    for (int j = 0; j < 4; ++j) {
      int uu = u + j;
      int lane = (uu & 15) + lhalf;
      size_t dst = ((((size_t)e * KB1 + kb) * NCT + (uu >> 4)) * 32 + lane) * 16 + s;
      W1b[dst] = f32_to_bf16(vv[j]);
    }
  } else if (gid < n1 + n2) {
    int idx = (gid - n1) * 4;
    int u = idx % UDIM; int rest = idx / UDIM;
    int k = rest % UDIM; int e = rest / UDIM;
    float4 v = *(const float4*)(proj + (size_t)idx);
    int kb = k >> 5, kr = k & 31;
    int s = (kr & 7) + ((kr >> 4) << 3), lhalf = ((kr >> 3) & 1) << 4;
    float vv[4] = {v.x, v.y, v.z, v.w};
#pragma unroll
    for (int j = 0; j < 4; ++j) {
      int uu = u + j;
      int lane = (uu & 15) + lhalf;
      size_t dst = ((((size_t)e * KB2 + kb) * NCT + (uu >> 4)) * 32 + lane) * 16 + s;
      projb[dst] = f32_to_bf16(vv[j]);
    }
  }
}

// ------------------------------------------------------------ gating (fp32, one wave32 per token)
__global__ __launch_bounds__(256) void gate_kernel(
    const float* __restrict__ x, const float* __restrict__ Wg,
    const float* __restrict__ bg, int* __restrict__ expert_id,
    int* __restrict__ counts) {
  int gtid = blockIdx.x * 256 + threadIdx.x;
  int tok  = gtid >> 5;
  int lane = gtid & 31;
  if (tok >= N_TOK) return;

  const float* xr = x + (size_t)tok * DIM;
  float acc[NEXP];
#pragma unroll
  for (int e = 0; e < NEXP; ++e) acc[e] = 0.f;

  for (int d = lane; d < DIM; d += 32) {
    float xv = xr[d];
    const float4* wr = (const float4*)(Wg + (size_t)d * NEXP);
    float4 w0 = wr[0], w1 = wr[1];
    acc[0] += xv * w0.x; acc[1] += xv * w0.y;
    acc[2] += xv * w0.z; acc[3] += xv * w0.w;
    acc[4] += xv * w1.x; acc[5] += xv * w1.y;
    acc[6] += xv * w1.z; acc[7] += xv * w1.w;
  }
#pragma unroll
  for (int off = 16; off > 0; off >>= 1) {
#pragma unroll
    for (int e = 0; e < NEXP; ++e) acc[e] += __shfl_xor(acc[e], off, 32);
  }
  if (lane == 0) {
    float best = acc[0] + bg[0]; int bi = 0;
#pragma unroll
    for (int e = 1; e < NEXP; ++e) {
      float v = acc[e] + bg[e];
      if (v > best) { best = v; bi = e; }   // strict > == first-max (jnp.argmax)
    }
    expert_id[tok] = bi;
    atomicAdd(&counts[bi], 1);
  }
}

// ------------------------------------------------------------ tile-offset scan
__global__ void scan_kernel(const int* __restrict__ counts, int* __restrict__ tileStart) {
  if (threadIdx.x == 0 && blockIdx.x == 0) {
    int run = 0;
#pragma unroll
    for (int e = 0; e < NEXP; ++e) { tileStart[e] = run; run += (counts[e] + MT - 1) / MT; }
    tileStart[NEXP] = run;
  }
}

// ------------------------------------------------------------ scatter tokens into expert tiles
__global__ void scatter_kernel(const int* __restrict__ expert_id,
                               const int* __restrict__ tileStart,
                               int* __restrict__ cursors, int* __restrict__ perm) {
  int n = blockIdx.x * blockDim.x + threadIdx.x;
  if (n >= N_TOK) return;
  int e = expert_id[n];
  int slot = atomicAdd(&cursors[e], 1);
  perm[tileStart[e] * MT + slot] = n;
}

// ------------------------------------------------------------ main expert kernel
// One workgroup (8 wave32) per 32-token tile (two 16-row m-tiles sharing B fragments).
// Inner loops are software-pipelined: ping-pong register buffers for the A (LDS)
// and B (global, pre-swizzled bf16) fragments, kb loop unrolled by 2.
__global__ __launch_bounds__(256) void moe_kernel(
    const float* __restrict__ x,
    const unsigned short* __restrict__ W1b, const float* __restrict__ b1,
    const unsigned short* __restrict__ projb,
    const float* __restrict__ ctrl, const float* __restrict__ scaling,
    const int* __restrict__ perm, const int* __restrict__ tileStart,
    float* __restrict__ out) {
  // A fragments in LDS (WMMA-A bf16 layout):
  // element (m,k): oct=(k%32)/8, lane=(m%16)+16*(oct&1), slot=(k%8)+8*(oct>>1)
  // GEMM1 frag base: ((mt*KB1 + kb)*32 + lane)*16 ; GEMM2 (H): ((mt*KB2 + kb)*32 + lane)*16
  __shared__ __attribute__((aligned(32))) unsigned short Abuf[2 * KB1 * 32 * 16]; // 64 KB
  __shared__ int rows[MT];

  const int t = blockIdx.x;
  const int totalTiles = tileStart[NEXP];
  if (t >= totalTiles) return;            // uniform exit; EXEC all-ones for WMMA

  int e = 0;
#pragma unroll
  for (int k = 1; k < NEXP; ++k) if (t >= tileStart[k]) e = k;

  const int tid  = threadIdx.x;
  const int lane = tid & 31;
  const int wv   = tid >> 5;
  const int col0 = lane & 15;
  const int mBase = (lane >> 4) << 3;     // C frag: lanes 0-15 rows 0-7, 16-31 rows 8-15

  if (tid < MT) rows[tid] = perm[t * MT + tid];
  __syncthreads();

  // ---- stage A: x rows (32 x 1024 f32) -> bf16 swizzled into LDS ----
  {
    const int ml = tid >> 4;              // 0..15 row within half
    const int c4 = (tid & 15) << 2;
    for (int pass = 0; pass < 32; ++pass) {
      int m = ml + ((pass & 1) << 4);     // alternate row halves
      int d = (pass >> 1) * 64 + c4;
      int r = rows[m];
      float4 v;
      if (r >= 0) v = *(const float4*)(x + (size_t)r * DIM + d);
      else        v = make_float4(0.f, 0.f, 0.f, 0.f);
      int kb = d >> 5, rr = d & 31, oct = rr >> 3;
      int ln = (m & 15) + ((oct & 1) << 4);
      int s  = (rr & 7) + ((oct >> 1) << 3);
      int mt = m >> 4;
      uint2 p;
      p.x = (unsigned)f32_to_bf16(v.x) | ((unsigned)f32_to_bf16(v.y) << 16);
      p.y = (unsigned)f32_to_bf16(v.z) | ((unsigned)f32_to_bf16(v.w) << 16);
      *(uint2*)&Abuf[(((mt * KB1 + kb) * 32 + ln) << 4) + s] = p;
    }
  }
  __syncthreads();

  // ---- GEMM1: H = swish(A @ W1[e] + b1[e]),  32x512, K=1024 ----
  v8f acc[2][4] = {};
  {
    const unsigned short* btBase =
        W1b + (size_t)e * KB1 * KSTR + ((size_t)(wv << 2) * 32 + lane) * 16;
    v16bf afb[2][2], bvb[2][4];
    afb[0][0] = *(const v16bf*)&Abuf[(lane) << 4];
    afb[0][1] = *(const v16bf*)&Abuf[((KB1 * 32) + lane) << 4];
#pragma unroll
    for (int ci = 0; ci < 4; ++ci)
      bvb[0][ci] = *(const v16bf*)(btBase + ci * (32 * 16));
#pragma unroll 2
    for (int kb = 0; kb < KB1; ++kb) {
      const int cur = kb & 1, nxt = cur ^ 1;
      if (kb + 1 < KB1) {
        const unsigned short* btn = btBase + (size_t)(kb + 1) * KSTR;
        __builtin_prefetch(btn + KSTR, 0, 1);
#pragma unroll
        for (int ci = 0; ci < 4; ++ci)
          bvb[nxt][ci] = *(const v16bf*)(btn + ci * (32 * 16));
        afb[nxt][0] = *(const v16bf*)&Abuf[((kb + 1) * 32 + lane) << 4];
        afb[nxt][1] = *(const v16bf*)&Abuf[(((KB1 + kb + 1) * 32) + lane) << 4];
      }
#pragma unroll
      for (int ci = 0; ci < 4; ++ci) {
        acc[0][ci] = __builtin_amdgcn_wmma_f32_16x16x32_bf16(
            false, afb[cur][0], false, bvb[cur][ci], (short)0, acc[0][ci], false, false);
        acc[1][ci] = __builtin_amdgcn_wmma_f32_16x16x32_bf16(
            false, afb[cur][1], false, bvb[cur][ci], (short)0, acc[1][ci], false, false);
      }
    }
  }

  __syncthreads();   // all waves done reading A before H overwrites Abuf

  // ---- bias + swish, write H (32x512) back to LDS in WMMA-A layout ----
  {
    const float* b1e = b1 + (size_t)e * UDIM;
#pragma unroll
    for (int ci = 0; ci < 4; ++ci) {
      int u = (wv * 4 + ci) * 16 + col0;
      float bias = b1e[u];
      int kb2 = u >> 5, rr = u & 31, oct = rr >> 3;
      int s2  = (rr & 7) + ((oct >> 1) << 3);
      int lnA = (oct & 1) << 4;
#pragma unroll
      for (int mt = 0; mt < 2; ++mt) {
#pragma unroll
        for (int vr = 0; vr < 8; ++vr) {
          int ml = mBase + vr;
          float tv = acc[mt][ci][vr] + bias;
          float hs = tv / (1.f + __expf(-tv));          // swish
          Abuf[(((mt * KB2 + kb2) * 32 + (ml + lnA)) << 4) + s2] = f32_to_bf16(hs);
        }
      }
    }
  }
  __syncthreads();

  // ---- GEMM2: Z = H @ proj[e],  32x512, K=512 ----
  v8f zacc[2][4] = {};
  {
    const unsigned short* btBase =
        projb + (size_t)e * KB2 * KSTR + ((size_t)(wv << 2) * 32 + lane) * 16;
    v16bf afb[2][2], bvb[2][4];
    afb[0][0] = *(const v16bf*)&Abuf[(lane) << 4];
    afb[0][1] = *(const v16bf*)&Abuf[((KB2 * 32) + lane) << 4];
#pragma unroll
    for (int ci = 0; ci < 4; ++ci)
      bvb[0][ci] = *(const v16bf*)(btBase + ci * (32 * 16));
#pragma unroll 2
    for (int kb = 0; kb < KB2; ++kb) {
      const int cur = kb & 1, nxt = cur ^ 1;
      if (kb + 1 < KB2) {
        const unsigned short* btn = btBase + (size_t)(kb + 1) * KSTR;
        __builtin_prefetch(btn + KSTR, 0, 1);
#pragma unroll
        for (int ci = 0; ci < 4; ++ci)
          bvb[nxt][ci] = *(const v16bf*)(btn + ci * (32 * 16));
        afb[nxt][0] = *(const v16bf*)&Abuf[((kb + 1) * 32 + lane) << 4];
        afb[nxt][1] = *(const v16bf*)&Abuf[(((KB2 + kb + 1) * 32) + lane) << 4];
      }
#pragma unroll
      for (int ci = 0; ci < 4; ++ci) {
        zacc[0][ci] = __builtin_amdgcn_wmma_f32_16x16x32_bf16(
            false, afb[cur][0], false, bvb[cur][ci], (short)0, zacc[0][ci], false, false);
        zacc[1][ci] = __builtin_amdgcn_wmma_f32_16x16x32_bf16(
            false, afb[cur][1], false, bvb[cur][ci], (short)0, zacc[1][ci], false, false);
      }
    }
  }

  // ---- epilogue: sigmoid -> normalized Gaussian RBF -> ctrl dot -> scale -> scatter ----
  const float* Ce = ctrl + (size_t)e * NBASE * UDIM;
  const float* Se = scaling + (size_t)e * UDIM;
  const float knotStep = 1.0f / (float)(NBASE - 1);    // linspace(0,1,8)
#pragma unroll
  for (int ci = 0; ci < 4; ++ci) {
    int u = (wv * 4 + ci) * 16 + col0;
    float sc = Se[u];
    float cb[NBASE];
#pragma unroll
    for (int b = 0; b < NBASE; ++b) cb[b] = Ce[b * UDIM + u];
#pragma unroll
    for (int mt = 0; mt < 2; ++mt) {
#pragma unroll
      for (int vr = 0; vr < 8; ++vr) {
        int m = mt * 16 + mBase + vr;
        int r = rows[m];
        float z  = zacc[mt][ci][vr];
        float xn = 1.f / (1.f + __expf(-z));
        float bs = 0.f, dt = 0.f;
#pragma unroll
        for (int b = 0; b < NBASE; ++b) {
          float dd = xn - (float)b * knotStep;
          float w  = __expf(-32.f * dd * dd);          // 1/(2*(1/8)^2) = 32
          bs += w; dt += w * cb[b];
        }
        float val = dt / (bs + 1e-6f) * sc;
        if (r >= 0) out[(size_t)r * UDIM + u] = val;
      }
    }
  }
}

// ------------------------------------------------------------ launcher
extern "C" void kernel_launch(void* const* d_in, const int* in_sizes, int n_in,
                              void* d_out, int out_size, void* d_ws, size_t ws_size,
                              hipStream_t stream) {
  const float* x    = (const float*)d_in[0];
  const float* W1   = (const float*)d_in[1];
  const float* b1   = (const float*)d_in[2];
  const float* proj = (const float*)d_in[3];
  const float* ctrl = (const float*)d_in[4];
  const float* scl  = (const float*)d_in[5];
  const float* Wg   = (const float*)d_in[6];
  const float* bg   = (const float*)d_in[7];
  float* out = (float*)d_out;

  // workspace: bf16 weight tiles first (alignment), then routing ints
  unsigned short* W1b   = (unsigned short*)d_ws;                 // E*D*U   = 4M bf16
  unsigned short* projb = W1b + (size_t)NEXP * DIM * UDIM;       // E*U*U   = 2M bf16
  int* wsI       = (int*)(projb + (size_t)NEXP * UDIM * UDIM);
  int* expert_id = wsI;                      // N
  int* counts    = wsI + N_TOK;              // E
  int* cursors   = counts + NEXP;            // E
  int* tileStart = cursors + NEXP;           // E+1
  int* perm      = tileStart + (NEXP + 1);   // MAXT*MT

  const int nconv = (NEXP * DIM * UDIM + NEXP * UDIM * UDIM) / 4;

  init_kernel<<<(MAXT * MT + 255) / 256, 256, 0, stream>>>(counts, cursors, perm);
  wconv_kernel<<<(nconv + 255) / 256, 256, 0, stream>>>(W1, proj, W1b, projb);
  gate_kernel<<<(N_TOK * 32) / 256, 256, 0, stream>>>(x, Wg, bg, expert_id, counts);
  scan_kernel<<<1, 32, 0, stream>>>(counts, tileStart);
  scatter_kernel<<<(N_TOK + 255) / 256, 256, 0, stream>>>(expert_id, tileStart, cursors, perm);
  moe_kernel<<<MAXT, 256, 0, stream>>>(x, W1b, b1, projb, ctrl, scl, perm, tileStart, out);
}
